// TreeResUNet_45664092291379
// MI455X (gfx1250) — matrix operand order repaired and checked by reading, not compiled, not run on hardware
//
#include <hip/hip_runtime.h>

// ---------------------------------------------------------------------------
// MI455X (gfx1250, wave32, WMMA) implementation of TreeResUNet forward.
// Dominant compute (85+ GFLOP of serial ring-LSTM GEMMs + 28 GFLOP of
// down-transforms) runs on v_wmma_f32_16x16x32_f16 with LDS double-buffered
// weight tiles. Elementwise/norm stages are wave32 VALU kernels.
// ---------------------------------------------------------------------------

typedef __attribute__((ext_vector_type(16))) _Float16 v16h;
typedef __attribute__((ext_vector_type(8)))  _Float16 v8h;
typedef __attribute__((ext_vector_type(8)))  float    v8f;

#define NN   16384   // N_NODES
#define LVD  8       // L_LEVELS
#define WVD  2048    // W_LEVEL
#define HSZ  192     // H_SIZE

static __device__ __forceinline__ float sigf(float v) { return 1.f / (1.f + __expf(-v)); }

// ---------------------------------------------------------------------------
// f32 -> f16 with K zero-padding to Kp (Kp % 32 == 0 for WMMA K-steps)
// ---------------------------------------------------------------------------
__global__ void k_f32_to_f16_pad(const float* __restrict__ src, _Float16* __restrict__ dst,
                                 int rows, int K, int Kp) {
  long i = (long)blockIdx.x * blockDim.x + threadIdx.x;
  if (i >= (long)rows * Kp) return;
  int  c = (int)(i % Kp);
  long r = i / Kp;
  dst[i] = (c < K) ? (_Float16)src[r * (long)K + c] : (_Float16)0.f;
}

__global__ void k_zero(float* __restrict__ p, long n) {
  long i = (long)blockIdx.x * blockDim.x + threadIdx.x;
  if (i < n) p[i] = 0.f;
}

// ---------------------------------------------------------------------------
// WMMA GEMM:  C[M,N] = act( A[M,Kp](f16) * W[N,Kp](f16)^T + bias[N] )
//   block = 256 threads = 8 waves; block tile 128(M) x 64(N)
//   wave tile 16(M) x 64(N): 1 A-fragment x 4 B-fragments per K-step
//   W tile (64 rows x 32 k, f16) double-buffered in LDS
//   act: 0 = none, 1 = sigmoid
// ---------------------------------------------------------------------------
#define LWS 40  // padded LDS row stride in halfs (32 k + 8 pad)

__global__ __launch_bounds__(256, 2) void k_gemm_wmma(
    const _Float16* __restrict__ A, const _Float16* __restrict__ W,
    const float* __restrict__ bias, float* __restrict__ C,
    int M, int N, int Kp, int act) {
  __shared__ _Float16 lw[2][64 * LWS];

  const int tid  = threadIdx.x;
  const int lane = tid & 31;
  const int wave = tid >> 5;
  const int nb   = N >> 6;                     // N / 64
  const int n0   = (blockIdx.x % nb) * 64;
  const int m0   = (blockIdx.x / nb) * 128 + wave * 16;

  // A fragment addressing (ISA 16-bit A 16x32 layout)
  const int arow  = m0 + (lane & 15);
  const int akoff = (lane >> 4) * 8;           // lanes 16-31 start at K=8
  const _Float16* aptr = A + (size_t)arow * Kp + akoff;
  // B fragment addressing (ISA 16-bit B 32x16 layout): lanes 16-31 start at K=8
  const int bkoff = (lane >> 4) * 8;

  v8f acc[4];
#pragma unroll
  for (int j = 0; j < 4; ++j) {
    float b = bias ? bias[n0 + j * 16 + (lane & 15)] : 0.f;
#pragma unroll
    for (int r = 0; r < 8; ++r) acc[j][r] = b;
  }

  auto stage = [&](int buf, int k0) {
    // 256 threads cooperatively load 64 rows x 32 halfs (8 halfs each)
    int row = tid >> 2;
    int seg = (tid & 3) * 8;
    v8h v = *(const v8h*)(W + (size_t)(n0 + row) * Kp + k0 + seg);
    *(v8h*)(&lw[buf][row * LWS + seg]) = v;
  };

  stage(0, 0);
  const int ksteps = Kp >> 5;
  for (int kt = 0; kt < ksteps; ++kt) {
    __syncthreads();                            // staged tile ready
    if (kt + 1 < ksteps) stage((kt + 1) & 1, (kt + 1) << 5);

    const _Float16* ap = aptr + ((size_t)kt << 5);
    if (kt + 1 < ksteps) __builtin_prefetch(ap + 32, 0, 1);   // global_prefetch_b8

    v16h afrag;
    {
      v8h* ah = (v8h*)&afrag;
      ah[0] = *(const v8h*)(ap);                // K 0-7  (or 8-15)
      ah[1] = *(const v8h*)(ap + 16);           // K 16-23(or 24-31)
    }
    const _Float16* lb = lw[kt & 1];
#pragma unroll
    for (int j = 0; j < 4; ++j) {
      v16h bfrag;
      const _Float16* bp = lb + (j * 16 + (lane & 15)) * LWS + bkoff;
      v8h* bh = (v8h*)&bfrag;
      bh[0] = *(const v8h*)(bp);                // K 0-7  (or 8-15)
      bh[1] = *(const v8h*)(bp + 16);           // K 16-23(or 24-31)
      acc[j] = __builtin_amdgcn_wmma_f32_16x16x32_f16(
          false, afrag, false, bfrag, (short)0, acc[j], false, false);
    }
  }

#pragma unroll
  for (int j = 0; j < 4; ++j)
#pragma unroll
    for (int r = 0; r < 8; ++r) {
      int mm = m0 + r + ((lane >> 4) << 3);
      int nn = n0 + j * 16 + (lane & 15);
      float v = acc[j][r];
      if (act == 1) v = sigf(v);
      C[(size_t)mm * N + nn] = v;
    }
}

// ---------------------------------------------------------------------------
// Row-wise LayerNorm (wave per row), optional ReLU, in-place safe
// ---------------------------------------------------------------------------
__global__ void k_layernorm(const float* __restrict__ x, const float* __restrict__ g,
                            const float* __restrict__ b, float* __restrict__ y,
                            long rows, int D, int relu) {
  long row = (long)blockIdx.x * (blockDim.x >> 5) + (threadIdx.x >> 5);
  int lane = threadIdx.x & 31;
  if (row >= rows) return;
  const float* xr = x + row * D;
  float s = 0.f, s2 = 0.f;
  for (int c = lane; c < D; c += 32) { float v = xr[c]; s += v; s2 += v * v; }
  for (int o = 16; o; o >>= 1) { s += __shfl_xor(s, o, 32); s2 += __shfl_xor(s2, o, 32); }
  float m = s / D, var = s2 / D - m * m, rs = rsqrtf(var + 1e-5f);
  float* yr = y + row * D;
  for (int c = lane; c < D; c += 32) {
    float v = (xr[c] - m) * rs * g[c] + b[c];
    if (relu) v = fmaxf(v, 0.f);
    yr[c] = v;
  }
}

// ---------------------------------------------------------------------------
// InstanceNorm over width for channel slice [c0,c0+Cc) of (N,Ct,Wd) buffer
// ---------------------------------------------------------------------------
__global__ void k_inorm(float* __restrict__ buf, int Ct, int c0, int Cc,
                        int N, int Wd, int relu) {
  long row = (long)blockIdx.x * (blockDim.x >> 5) + (threadIdx.x >> 5);
  int lane = threadIdx.x & 31;
  if (row >= (long)N * Cc) return;
  long n = row / Cc; int c = c0 + (int)(row % Cc);
  float* p = buf + (n * Ct + c) * (long)Wd;
  float s = 0.f, s2 = 0.f;
  for (int w = lane; w < Wd; w += 32) { float v = p[w]; s += v; s2 += v * v; }
  for (int o = 16; o; o >>= 1) { s += __shfl_xor(s, o, 32); s2 += __shfl_xor(s2, o, 32); }
  float m = s / Wd, var = s2 / Wd - m * m, rs = rsqrtf(var + 1e-5f);
  for (int w = lane; w < Wd; w += 32) {
    float v = (p[w] - m) * rs;
    if (relu) v = fmaxf(v, 0.f);
    p[w] = v;
  }
}

// ---------------------------------------------------------------------------
// 1x3 conv (zero pad), slice-to-slice, optional residual slice
// ---------------------------------------------------------------------------
__global__ void k_conv1x3(const float* __restrict__ x, int xCt, int xc0,
                          const float* __restrict__ wgt, int Cin, int Cout,
                          const float* __restrict__ res, int rCt, int rc0,
                          float* __restrict__ y, int yCt, int yc0,
                          int N, int Wd) {
  long i = (long)blockIdx.x * blockDim.x + threadIdx.x;
  if (i >= (long)N * Cout * Wd) return;
  int  w  = (int)(i % Wd);
  int  co = (int)((i / Wd) % Cout);
  long n  = i / ((long)Wd * Cout);
  float s = 0.f;
  for (int ci = 0; ci < Cin; ++ci) {
    const float* xr = x + (n * xCt + xc0 + ci) * (long)Wd;
    const float* wr = wgt + ((long)co * Cin + ci) * 3;
    if (w > 0)      s += xr[w - 1] * wr[0];
    s += xr[w] * wr[1];
    if (w < Wd - 1) s += xr[w + 1] * wr[2];
  }
  if (res) s += res[(n * rCt + rc0 + co) * (long)Wd + w];
  y[(n * yCt + yc0 + co) * (long)Wd + w] = s;
}

__global__ void k_maxpool2(const float* __restrict__ src, int C,
                           float* __restrict__ dst, int N, int Win, int relu) {
  int Wout = Win >> 1;
  long i = (long)blockIdx.x * blockDim.x + threadIdx.x;
  if (i >= (long)N * C * Wout) return;
  int  w  = (int)(i % Wout);
  long nc = i / Wout;
  float a = src[nc * Win + 2 * w], b = src[nc * Win + 2 * w + 1];
  float v = fmaxf(a, b);
  if (relu) v = fmaxf(v, 0.f);
  dst[i] = v;
}

__global__ void k_copy_slice(const float* __restrict__ src, int sCt, int sc0,
                             float* __restrict__ dst, int dCt, int dc0,
                             int Cc, int N, int Wd, int relu) {
  long i = (long)blockIdx.x * blockDim.x + threadIdx.x;
  if (i >= (long)N * Cc * Wd) return;
  int  w = (int)(i % Wd);
  int  c = (int)((i / Wd) % Cc);
  long n = i / ((long)Wd * Cc);
  float v = src[(n * sCt + sc0 + c) * (long)Wd + w];
  if (relu) v = fmaxf(v, 0.f);
  dst[(n * dCt + dc0 + c) * (long)Wd + w] = v;
}

// bicubic x2 upsample (a=-0.75), clamped taps, per (n,c) row
__global__ void k_bicubic_up(const float* __restrict__ x, int C, int N, int Win,
                             float* __restrict__ y) {
  int Wout = 2 * Win;
  long i = (long)blockIdx.x * blockDim.x + threadIdx.x;
  if (i >= (long)N * C * Wout) return;
  int  wo = (int)(i % Wout);
  long nc = i / Wout;
  float xx = (float)wo * (float)(Win - 1) / (float)(Wout - 1);
  int   x0 = (int)floorf(xx);
  float t  = xx - (float)x0;
  const float a = -0.75f;
  float tp = t + 1.f, tm = 1.f - t, tq = 2.f - t;
  float wm = a*tp*tp*tp - 5.f*a*tp*tp + 8.f*a*tp - 4.f*a;
  float w0 = (a+2.f)*t*t*t - (a+3.f)*t*t + 1.f;
  float w1 = (a+2.f)*tm*tm*tm - (a+3.f)*tm*tm + 1.f;
  float w2 = a*tq*tq*tq - 5.f*a*tq*tq + 8.f*a*tq - 4.f*a;
  const float* xr = x + nc * Win;
  int j0 = x0 - 1; if (j0 < 0) j0 = 0; if (j0 > Win - 1) j0 = Win - 1;
  int j1 = x0;     if (j1 < 0) j1 = 0; if (j1 > Win - 1) j1 = Win - 1;
  int j2 = x0 + 1; if (j2 < 0) j2 = 0; if (j2 > Win - 1) j2 = Win - 1;
  int j3 = x0 + 2; if (j3 < 0) j3 = 0; if (j3 > Win - 1) j3 = Win - 1;
  y[i] = wm * xr[j0] + w0 * xr[j1] + w1 * xr[j2] + w2 * xr[j3];
}

// small dense (h_mlp): thread per (row, out)
__global__ void k_dense_small(const float* __restrict__ x, const float* __restrict__ w,
                              const float* __restrict__ b, float* __restrict__ y,
                              int rows, int K, int Nc) {
  long i = (long)blockIdx.x * blockDim.x + threadIdx.x;
  if (i >= (long)rows * Nc) return;
  int  o = (int)(i % Nc);
  long r = i / Nc;
  const float* xr = x + r * K;
  const float* wr = w + (long)o * K;
  float s = b ? b[o] : 0.f;
  for (int k = 0; k < K; ++k) s += xr[k] * wr[k];
  y[i] = s;
}

// ---------------------------------------------------------------------------
// tree-LSTM helpers
// ---------------------------------------------------------------------------
// gather ring neighbors: hcat = [h[w], h[w+dir]], cpair likewise (192 each)
__global__ void k_gather_ring(const float* __restrict__ hsrc, const float* __restrict__ csrc,
                              _Float16* __restrict__ hcat16, float* __restrict__ hcat32,
                              float* __restrict__ cpair, int Wv, int dir) {
  int w = blockIdx.x, t = threadIdx.x;      // blockDim = 192
  int n1 = (w + dir + Wv) % Wv;
  float h0v = hsrc[(size_t)w  * HSZ + t];
  float h1v = hsrc[(size_t)n1 * HSZ + t];
  hcat32[(size_t)w * 384 + t]       = h0v;
  hcat32[(size_t)w * 384 + 192 + t] = h1v;
  hcat16[(size_t)w * 384 + t]       = (_Float16)h0v;
  hcat16[(size_t)w * 384 + 192 + t] = (_Float16)h1v;
  cpair[(size_t)w * 384 + t]        = csrc[(size_t)w  * HSZ + t];
  cpair[(size_t)w * 384 + 192 + t]  = csrc[(size_t)n1 * HSZ + t];
}

// c_sum = f[:,0,:]*c[:,0,:] + f[:,1,:]*c[:,1,:]   (f already sigmoid'ed)
__global__ void k_gate_csum(const float* __restrict__ f, const float* __restrict__ cpair,
                            float* __restrict__ csum, int Wv) {
  long i = (long)blockIdx.x * blockDim.x + threadIdx.x;
  if (i >= (long)Wv * HSZ) return;
  int  t = (int)(i % HSZ);
  long w = i / HSZ;
  csum[i] = f[w * 384 + t] * cpair[w * 384 + t] +
            f[w * 384 + 192 + t] * cpair[w * 384 + 192 + t];
}

// msg_norm -> f16 GEMM operand: out = msg/||msg|| * ||x|| * s   (wave per row)
__global__ void k_msgnorm_f16(const float* __restrict__ msg, int Dm,
                              const float* __restrict__ x, int Dx, int xstride,
                              const float* __restrict__ s,
                              _Float16* __restrict__ out, int rows) {
  int row = blockIdx.x * (blockDim.x >> 5) + (threadIdx.x >> 5);
  int lane = threadIdx.x & 31;
  if (row >= rows) return;
  const float* mr = msg + (size_t)row * Dm;
  const float* xr = x + (size_t)row * xstride;
  float sm = 0.f, sx = 0.f;
  for (int c = lane; c < Dm; c += 32) { float v = mr[c]; sm += v * v; }
  for (int c = lane; c < Dx; c += 32) { float v = xr[c]; sx += v * v; }
  for (int o = 16; o; o >>= 1) { sm += __shfl_xor(sm, o, 32); sx += __shfl_xor(sx, o, 32); }
  float k = sqrtf(sx) * s[0] / fmaxf(sqrtf(sm), 1e-12f);
  _Float16* orow = out + (size_t)row * Dm;
  for (int c = lane; c < Dm; c += 32) orow[c] = (_Float16)(mr[c] * k);
}

__global__ void k_msgnorm_f32(const float* __restrict__ msg, int Dm,
                              const float* __restrict__ x, int Dx, int xstride,
                              const float* __restrict__ s,
                              float* __restrict__ out, int rows) {
  int row = blockIdx.x * (blockDim.x >> 5) + (threadIdx.x >> 5);
  int lane = threadIdx.x & 31;
  if (row >= rows) return;
  const float* mr = msg + (size_t)row * Dm;
  const float* xr = x + (size_t)row * xstride;
  float sm = 0.f, sx = 0.f;
  for (int c = lane; c < Dm; c += 32) { float v = mr[c]; sm += v * v; }
  for (int c = lane; c < Dx; c += 32) { float v = xr[c]; sx += v * v; }
  for (int o = 16; o; o >>= 1) { sm += __shfl_xor(sm, o, 32); sx += __shfl_xor(sx, o, 32); }
  float k = sqrtf(sx) * s[0] / fmaxf(sqrtf(sm), 1e-12f);
  float* orow = out + (size_t)row * Dm;
  for (int c = lane; c < Dm; c += 32) orow[c] = mr[c] * k;
}

// apply node: i,o,u = split(iou + b); c = relu(i)*tanh(u) + c_in; h = sig(o)*tanh(c)
__global__ void k_apply_node(const float* __restrict__ iou, const float* __restrict__ biou,
                             const float* __restrict__ c_in, float* __restrict__ h,
                             float* __restrict__ c_out, int Wv) {
  long i = (long)blockIdx.x * blockDim.x + threadIdx.x;
  if (i >= (long)Wv * HSZ) return;
  int  t = (int)(i % HSZ);
  long w = i / HSZ;
  const float* ir = iou + w * 576;
  float iv = ir[t]       + biou[t];
  float ov = ir[192 + t] + biou[192 + t];
  float uv = ir[384 + t] + biou[384 + t];
  float c  = fmaxf(iv, 0.f) * tanhf(uv) + c_in[i];
  c_out[i] = c;
  h[i]     = sigf(ov) * tanhf(c);
}

// vs_pre = concat(hN (192), iouN (576)) per row
__global__ void k_pack_vs(const float* __restrict__ h, const float* __restrict__ iou,
                          float* __restrict__ out, int N) {
  long i = (long)blockIdx.x * blockDim.x + threadIdx.x;
  if (i >= (long)N * 768) return;
  int  t = (int)(i % 768);
  long n = i / 768;
  out[i] = (t < 192) ? h[n * HSZ + t] : iou[n * 576 + (t - 192)];
}

// fused d0 = inorm(concat(xs0,v0,rup)) ; out[n,w] = sum_c d0[n,c,w]*ow[c] + ob
// one wave per n; each lane covers w and w+32
__global__ void k_out_fused(const float* __restrict__ xs0, const float* __restrict__ v0,
                            const float* __restrict__ rup, const float* __restrict__ ow,
                            const float* __restrict__ ob, float* __restrict__ out, int N) {
  int n = blockIdx.x, lane = threadIdx.x;     // blockDim = 32
  float acc0 = ob[0], acc1 = ob[0];
  for (int c = 0; c < 43; ++c) {
    const float* src; int ct, cl;
    if (c < 10)      { src = xs0; ct = 10; cl = c; }
    else if (c < 34) { src = v0;  ct = 24; cl = c - 10; }
    else             { src = rup; ct = 9;  cl = c - 34; }
    const float* p = src + ((size_t)n * ct + cl) * 64;
    float a = p[lane], b = p[lane + 32];
    float s = a + b, s2 = a * a + b * b;
    for (int o = 16; o; o >>= 1) { s += __shfl_xor(s, o, 32); s2 += __shfl_xor(s2, o, 32); }
    float m = s / 64.f, var = s2 / 64.f - m * m, rs = rsqrtf(var + 1e-5f);
    float wc = ow[c];
    acc0 += (a - m) * rs * wc;
    acc1 += (b - m) * rs * wc;
  }
  out[(size_t)n * 64 + lane]      = acc0;
  out[(size_t)n * 64 + lane + 32] = acc1;
}

// ---------------------------------------------------------------------------
// host helpers
// ---------------------------------------------------------------------------
static inline unsigned cdiv(long a, int b) { return (unsigned)((a + b - 1) / b); }

static void conv_h(const float* x, int xCt, int xc0, const float* w, int Cin, int Cout,
                   const float* res, int rCt, int rc0, float* y, int yCt, int yc0,
                   int N, int Wd, hipStream_t st) {
  long tot = (long)N * Cout * Wd;
  k_conv1x3<<<cdiv(tot, 256), 256, 0, st>>>(x, xCt, xc0, w, Cin, Cout, res, rCt, rc0,
                                            y, yCt, yc0, N, Wd);
}
static void inorm_h(float* buf, int Ct, int c0, int Cc, int N, int Wd, int relu,
                    hipStream_t st) {
  long rows = (long)N * Cc;
  k_inorm<<<cdiv(rows, 8), 256, 0, st>>>(buf, Ct, c0, Cc, N, Wd, relu);
}
static void res1d_h(const float* src, int sCt, int sc0, int Cin0,
                    const float* const* w, int nw, int Cout,
                    float* y, int yCt, int N, int Wd, hipStream_t st) {
  conv_h(src, sCt, sc0, w[0], Cin0, Cout, nullptr, 0, 0, y, yCt, 0, N, Wd, st);
  inorm_h(y, yCt, 0, Cout, N, Wd, 0, st);
  for (int k = 1; k < nw; ++k) {
    conv_h(y, yCt, (k - 1) * Cout, w[k], Cout, Cout, y, yCt, (k - 1) * Cout,
           y, yCt, k * Cout, N, Wd, st);
    inorm_h(y, yCt, k * Cout, Cout, N, Wd, 0, st);
  }
}
static void gemm_h(const _Float16* A, const _Float16* W, const float* bias, float* C,
                   int M, int N, int Kp, int act, hipStream_t st) {
  k_gemm_wmma<<<dim3((M / 128) * (N / 64)), 256, 0, st>>>(A, W, bias, C, M, N, Kp, act);
}
static void to16_h(const float* s, _Float16* d, int rows, int K, int Kp, hipStream_t st) {
  k_f32_to_f16_pad<<<cdiv((long)rows * Kp, 256), 256, 0, st>>>(s, d, rows, K, Kp);
}

// ---------------------------------------------------------------------------
extern "C" void kernel_launch(void* const* d_in, const int* in_sizes, int n_in,
                              void* d_out, int out_size, void* d_ws, size_t ws_size,
                              hipStream_t stream) {
  if (n_in < 89) return;  // expect x, h + 87 param leaves (jax tree order, dicts sorted)

  const float* X  = (const float*)d_in[0];   // (16384, 64)
  const float* H4 = (const float*)d_in[1];   // (16384, 4)
  auto P = [&](int i) { return (const float*)d_in[i]; };
  // leaf index map (alphabetical dict keys):
  //  2..4   c0 convs | 5..13 down_convs | 14+4i..17+4i down_transforms {b,be,g,w}
  //  30..39 h_mlp {b0,b1,b2,be0,be1,g0,g1,w0,w1,w2} | 40+2i,41+2i ls_norms {be,g}
  //  48+6i.. lstm {U_f_b,U_f_w,U_iou_w,b_iou,s_c,s_iou} | 72 out_b | 73 out_w
  //  74..77 up_convs | 78..88 up_lstm (u10,u21,u32,u43)

  // ---- workspace bump allocator ----
  char* wp = (char*)d_ws;
  auto alloc = [&](size_t bytes) -> void* {
    void* p = (void*)wp;
    wp += (bytes + 255) & ~(size_t)255;
    return p;
  };
  const int Kin[4] = {288, 432, 384, 384};
  const int Kpd[4] = {288, 448, 384, 384};

  _Float16 *Wdt16[4], *Uf16[4], *Uiou16[4];
  for (int i = 0; i < 4; ++i) Wdt16[i]  = (_Float16*)alloc((size_t)576 * Kpd[i] * 2);
  for (int i = 0; i < 4; ++i) Uf16[i]   = (_Float16*)alloc((size_t)384 * 384 * 2);
  for (int i = 0; i < 4; ++i) Uiou16[i] = (_Float16*)alloc((size_t)576 * 384 * 2);
  _Float16* A16big = (_Float16*)alloc((size_t)NN * 448 * 2);
  float* iouwork = (float*)alloc((size_t)NN * 576 * 4);
  float* h0 = (float*)alloc((size_t)NN * HSZ * 4);
  float* t1 = (float*)alloc((size_t)NN * 16 * 4);
  float* t2 = (float*)alloc((size_t)NN * 64 * 4);
  float* htree = (float*)alloc((size_t)NN * HSZ * 4);
  float* ctree = (float*)alloc((size_t)NN * HSZ * 4);
  _Float16* hcat16 = (_Float16*)alloc((size_t)WVD * 384 * 2);
  float* hcat32 = (float*)alloc((size_t)WVD * 384 * 4);
  float* cpair  = (float*)alloc((size_t)WVD * 384 * 4);
  float* fbuf   = (float*)alloc((size_t)WVD * 384 * 4);
  float* csum   = (float*)alloc((size_t)WVD * HSZ * 4);
  float* cnew   = (float*)alloc((size_t)WVD * HSZ * 4);
  _Float16* a16s = (_Float16*)alloc((size_t)WVD * 384 * 2);
  float* vsbuf = (float*)alloc((size_t)NN * 768 * 4);
  // encoder
  float* y0  = (float*)alloc((size_t)NN * 9 * 64 * 4);
  float* xs0 = (float*)alloc((size_t)NN * 10 * 64 * 4);
  float* xs1 = (float*)alloc((size_t)NN * 9 * 32 * 4);
  float* y1  = (float*)alloc((size_t)NN * 27 * 32 * 4);
  float* xs2 = (float*)alloc((size_t)NN * 27 * 16 * 4);
  float* y2  = (float*)alloc((size_t)NN * 48 * 16 * 4);
  float* xs3 = (float*)alloc((size_t)NN * 48 * 8 * 4);
  float* y3  = (float*)alloc((size_t)NN * 96 * 8 * 4);
  float* xs4 = (float*)alloc((size_t)NN * 96 * 4 * 4);
  // decoder seq temporaries
  float* v0b = (float*)alloc((size_t)NN * 24 * 64 * 4);
  float* tA  = (float*)alloc((size_t)NN * 24 * 64 * 4);
  float* tB  = (float*)alloc((size_t)NN * 24 * 32 * 4);
  float* tC  = (float*)alloc((size_t)NN * 24 * 32 * 4);
  float* tD  = (float*)alloc((size_t)NN * 24 * 16 * 4);
  float* tE  = (float*)alloc((size_t)NN * 24 * 16 * 4);
  float* v1b = (float*)alloc((size_t)NN * 24 * 32 * 4);
  float* v2b = (float*)alloc((size_t)NN * 24 * 16 * 4);
  float* v3b = (float*)alloc((size_t)NN * 24 * 8 * 4);
  float* ups  = (float*)alloc((size_t)NN * 2688 * 4);   // max upsample arena (42x64)
  float* d3b  = (float*)alloc((size_t)NN * 120 * 8 * 4);
  float* d2b  = (float*)alloc((size_t)NN * 78 * 16 * 4);
  float* d1b  = (float*)alloc((size_t)NN * 42 * 32 * 4);
  float* rbuf = (float*)alloc((size_t)NN * 9 * 64 * 4);
  (void)ws_size; (void)out_size; (void)in_sizes;

#define EL(kern, total, ...) kern<<<cdiv((long)(total), 256), 256, 0, stream>>>(__VA_ARGS__)

  // ---- weight conversion to f16 (K padded to mult of 32) ----
  for (int i = 0; i < 4; ++i) to16_h(P(17 + 4 * i), Wdt16[i], 576, Kin[i], Kpd[i], stream);
  for (int i = 0; i < 4; ++i) to16_h(P(49 + 6 * i), Uf16[i], 384, 384, 384, stream);
  for (int i = 0; i < 4; ++i) to16_h(P(50 + 6 * i), Uiou16[i], 576, 384, 384, stream);

  // ---- encoder: c0 res1d + down conv stacks ----
  {
    const float* w[3] = {P(2), P(3), P(4)};
    conv_h(X, 1, 0, w[0], 1, 3, nullptr, 0, 0, y0, 9, 0, NN, 64, stream);
    inorm_h(y0, 9, 0, 3, NN, 64, 0, stream);
    conv_h(y0, 9, 0, w[1], 3, 3, y0, 9, 0, y0, 9, 3, NN, 64, stream);
    inorm_h(y0, 9, 3, 3, NN, 64, 0, stream);
    conv_h(y0, 9, 3, w[2], 3, 3, y0, 9, 3, y0, 9, 6, NN, 64, stream);
    inorm_h(y0, 9, 6, 3, NN, 64, 0, stream);
  }
  EL(k_copy_slice, (long)NN * 9 * 64, y0, 9, 0, xs0, 10, 0, 9, NN, 64, 1);   // relu(y)
  EL(k_copy_slice, (long)NN * 1 * 64, X, 1, 0, xs0, 10, 9, 1, NN, 64, 0);    // raw x4
  EL(k_maxpool2, (long)NN * 9 * 32, y0, 9, xs1, NN, 64, 1);
  inorm_h(xs1, 9, 0, 9, NN, 32, 0, stream);
  { const float* w[3] = {P(5), P(6), P(7)};
    res1d_h(xs1, 9, 0, 9, w, 3, 9, y1, 27, NN, 32, stream); }
  EL(k_maxpool2, (long)NN * 27 * 16, y1, 27, xs2, NN, 32, 1);
  inorm_h(xs2, 27, 0, 27, NN, 16, 0, stream);
  { const float* w[3] = {P(8), P(9), P(10)};
    res1d_h(xs2, 27, 0, 27, w, 3, 16, y2, 48, NN, 16, stream); }
  EL(k_maxpool2, (long)NN * 48 * 8, y2, 48, xs3, NN, 16, 1);
  inorm_h(xs3, 48, 0, 48, NN, 8, 0, stream);
  { const float* w[3] = {P(11), P(12), P(13)};
    res1d_h(xs3, 48, 0, 48, w, 3, 32, y3, 96, NN, 8, stream); }
  EL(k_maxpool2, (long)NN * 96 * 4, y3, 96, xs4, NN, 8, 1);
  inorm_h(xs4, 96, 0, 96, NN, 4, 0, stream);

  // ---- h_mlp: 4 -> 16 -> 64 -> 192 (dense + LN + relu) ----
  EL(k_dense_small, (long)NN * 16, H4, P(37), P(30), t1, NN, 4, 16);
  k_layernorm<<<cdiv((long)NN, 8), 256, 0, stream>>>(t1, P(35), P(33), t1, NN, 16, 1);
  EL(k_dense_small, (long)NN * 64, t1, P(38), P(31), t2, NN, 16, 64);
  k_layernorm<<<cdiv((long)NN, 8), 256, 0, stream>>>(t2, P(36), P(34), t2, NN, 64, 1);
  EL(k_dense_small, (long)NN * 192, t2, P(39), P(32), h0, NN, 64, 192);

  const float* curp[4] = {xs1, xs2, xs3, xs4};

  // ---- per-scale: down transform (WMMA) + tree-LSTM + ls_norm + seq ----
  for (int ix = 0; ix < 4; ++ix) {
    to16_h(curp[ix], A16big, NN, Kin[ix], Kpd[ix], stream);
    gemm_h(A16big, Wdt16[ix], P(14 + 4 * ix), iouwork, NN, 576, Kpd[ix], 0, stream);
    k_layernorm<<<cdiv((long)NN, 8), 256, 0, stream>>>(
        iouwork, P(16 + 4 * ix), P(15 + 4 * ix), iouwork, NN, 576, 0);

    // tree prop over (8, 2048) ring levels
    const float* Ufb  = P(48 + 6 * ix);
    const float* biou = P(51 + 6 * ix);
    const float* sc   = P(52 + 6 * ix);
    const float* siou = P(53 + 6 * ix);
    hipMemcpyAsync(htree, h0, (size_t)NN * HSZ * 4, hipMemcpyDeviceToDevice, stream);
    EL(k_zero, (long)NN * HSZ, ctree, (long)NN * HSZ);

    auto level = [&](int l, int src, int dir) {
      float* hs  = htree + (size_t)src * WVD * HSZ;
      float* cs  = ctree + (size_t)src * WVD * HSZ;
      float* iol = iouwork + (size_t)l * WVD * 576;
      float* hl  = htree + (size_t)l * WVD * HSZ;
      float* cl  = ctree + (size_t)l * WVD * HSZ;
      k_gather_ring<<<WVD, 192, 0, stream>>>(hs, cs, hcat16, hcat32, cpair, WVD, dir);
      gemm_h(hcat16, Uf16[ix], Ufb, fbuf, WVD, 384, 384, /*sigmoid*/1, stream);
      EL(k_gate_csum, (long)WVD * HSZ, fbuf, cpair, csum, WVD);
      k_msgnorm_f16<<<cdiv(WVD, 8), 256, 0, stream>>>(hcat32, 384, iol, 576, 576,
                                                      siou, a16s, WVD);
      gemm_h(a16s, Uiou16[ix], nullptr, iol, WVD, 576, 384, 0, stream);
      k_msgnorm_f32<<<cdiv(WVD, 8), 256, 0, stream>>>(csum, HSZ, cpair, HSZ, 384,
                                                      sc, cnew, WVD);
      EL(k_apply_node, (long)WVD * HSZ, iol, biou, cnew, hl, cl, WVD);
    };
    EL(k_apply_node, (long)WVD * HSZ, iouwork, biou, ctree, htree, ctree, WVD);
    for (int l = 1; l < LVD; ++l) level(l, l - 1, +1);
    EL(k_apply_node, (long)WVD * HSZ, iouwork + (size_t)7 * WVD * 576, biou,
       ctree + (size_t)7 * WVD * HSZ, htree + (size_t)7 * WVD * HSZ,
       ctree + (size_t)7 * WVD * HSZ, WVD);
    for (int l = LVD - 2; l >= 0; --l) level(l, l + 1, -1);

    // vs = lnorm(concat(hN, iouN))
    EL(k_pack_vs, (long)NN * 768, htree, iouwork, vsbuf, NN);
    k_layernorm<<<cdiv((long)NN, 8), 256, 0, stream>>>(
        vsbuf, P(41 + 2 * ix), P(40 + 2 * ix), vsbuf, NN, 768, 0);

    // seq(): vs viewed as (N, 12, 64); up_lstm conv chains with pooling
    if (ix == 0) {
      const float* w[2] = {P(78), P(79)};
      res1d_h(vsbuf, 12, 0, 12, w, 2, 12, v0b, 24, NN, 64, stream);
      EL(k_copy_slice, (long)NN * 24 * 64, v0b, 24, 0, v0b, 24, 0, 24, NN, 64, 1);  // relu
    } else if (ix == 1) {
      const float* w[2] = {P(80), P(81)};
      res1d_h(vsbuf, 12, 0, 12, w, 2, 12, tA, 24, NN, 64, stream);
      EL(k_maxpool2, (long)NN * 24 * 32, tA, 24, v1b, NN, 64, 1);
    } else if (ix == 2) {
      const float* w0[2] = {P(82), P(83)};
      const float* w1[1] = {P(84)};
      res1d_h(vsbuf, 12, 0, 12, w0, 2, 12, tA, 24, NN, 64, stream);
      EL(k_maxpool2, (long)NN * 24 * 32, tA, 24, tB, NN, 64, 1);
      res1d_h(tB, 24, 0, 24, w1, 1, 24, tC, 24, NN, 32, stream);
      EL(k_maxpool2, (long)NN * 24 * 16, tC, 24, v2b, NN, 32, 1);
    } else {
      const float* w0[2] = {P(85), P(86)};
      const float* w1[1] = {P(87)};
      const float* w2[1] = {P(88)};
      res1d_h(vsbuf, 12, 0, 12, w0, 2, 12, tA, 24, NN, 64, stream);
      EL(k_maxpool2, (long)NN * 24 * 32, tA, 24, tB, NN, 64, 1);
      res1d_h(tB, 24, 0, 24, w1, 1, 24, tC, 24, NN, 32, stream);
      EL(k_maxpool2, (long)NN * 24 * 16, tC, 24, tD, NN, 32, 1);
      res1d_h(tD, 24, 0, 24, w2, 1, 24, tE, 24, NN, 16, stream);
      EL(k_maxpool2, (long)NN * 24 * 8, tE, 24, v3b, NN, 16, 1);
    }
  }

  // ---- decoder ----
  // d3 = inorm(concat(xs3(48), v3(24), res_up(xs4, up0 -> 48)))  width 8
  EL(k_copy_slice, (long)NN * 48 * 8, xs3, 48, 0, d3b, 120, 0, 48, NN, 8, 0);
  EL(k_copy_slice, (long)NN * 24 * 8, v3b, 24, 0, d3b, 120, 48, 24, NN, 8, 0);
  EL(k_bicubic_up, (long)NN * 96 * 8, xs4, 96, NN, 4, ups);
  conv_h(ups, 96, 0, P(74), 96, 48, nullptr, 0, 0, d3b, 120, 72, NN, 8, stream);
  inorm_h(d3b, 120, 72, 48, NN, 8, 1, stream);
  inorm_h(d3b, 120, 0, 120, NN, 8, 0, stream);
  // d2 = inorm(concat(xs2(27), v2(24), res_up(d3, up1 -> 27)))  width 16
  EL(k_copy_slice, (long)NN * 27 * 16, xs2, 27, 0, d2b, 78, 0, 27, NN, 16, 0);
  EL(k_copy_slice, (long)NN * 24 * 16, v2b, 24, 0, d2b, 78, 27, 24, NN, 16, 0);
  EL(k_bicubic_up, (long)NN * 120 * 16, d3b, 120, NN, 8, ups);
  conv_h(ups, 120, 0, P(75), 120, 27, nullptr, 0, 0, d2b, 78, 51, NN, 16, stream);
  inorm_h(d2b, 78, 51, 27, NN, 16, 1, stream);
  inorm_h(d2b, 78, 0, 78, NN, 16, 0, stream);
  // d1 = inorm(concat(xs1(9), v1(24), res_up(d2, up2 -> 9)))  width 32
  EL(k_copy_slice, (long)NN * 9 * 32, xs1, 9, 0, d1b, 42, 0, 9, NN, 32, 0);
  EL(k_copy_slice, (long)NN * 24 * 32, v1b, 24, 0, d1b, 42, 9, 24, NN, 32, 0);
  EL(k_bicubic_up, (long)NN * 78 * 32, d2b, 78, NN, 16, ups);
  conv_h(ups, 78, 0, P(76), 78, 9, nullptr, 0, 0, d1b, 42, 33, NN, 32, stream);
  inorm_h(d1b, 42, 33, 9, NN, 32, 1, stream);
  inorm_h(d1b, 42, 0, 42, NN, 32, 0, stream);
  // rup = res_up(d1, up3 -> 9) width 64; d0 inorm + projection fused into output
  EL(k_bicubic_up, (long)NN * 42 * 64, d1b, 42, NN, 32, ups);
  conv_h(ups, 42, 0, P(77), 42, 9, nullptr, 0, 0, rbuf, 9, 0, NN, 64, stream);
  inorm_h(rbuf, 9, 0, 9, NN, 64, 1, stream);
  k_out_fused<<<NN, 32, 0, stream>>>(xs0, v0b, rbuf, P(73), P(72), (float*)d_out, NN);

#undef EL
}
